// WindowAttention_77171972375038
// MI455X (gfx1250) — matrix operand order repaired and checked by reading, not compiled, not run on hardware
//
#include <hip/hip_runtime.h>
#include <hip/hip_bf16.h>

typedef __bf16 bf16_t;
typedef __attribute__((ext_vector_type(16))) __bf16 bf16x16;
typedef __attribute__((ext_vector_type(8)))  __bf16 bf16x8;
typedef __attribute__((ext_vector_type(4)))  __bf16 bf16x4;
typedef __attribute__((ext_vector_type(8)))  float  floatx8;

// ---- sizes ----
#define BWIN   4096
#define NTOK   64
#define CDIM   192
#define NHEAD  6
#define HD     32
#define NW     1024
#define QKVW_E (576 * 192)        // 110592
#define PROJW_E (192 * 192)       // 36864
#define BIAS_E (NHEAD * 64 * 64)  // 24576

__device__ __forceinline__ bf16x16 frag_2x8(const bf16_t* p0, const bf16_t* p1) {
  bf16x8 a = *(const bf16x8*)p0;
  bf16x8 b = *(const bf16x8*)p1;
  bf16x16 r;
#pragma unroll
  for (int i = 0; i < 8; ++i) { r[i] = a[i]; r[i + 8] = b[i]; }
  return r;
}

__device__ __forceinline__ bf16x16 frag_16(const bf16_t* p) {
  bf16x8 a = *(const bf16x8*)p;
  bf16x8 b = *(const bf16x8*)(p + 8);
  bf16x16 r;
#pragma unroll
  for (int i = 0; i < 8; ++i) { r[i] = a[i]; r[i + 8] = b[i]; }
  return r;
}

__device__ __forceinline__ floatx8 wmma_bf16(bf16x16 a, bf16x16 b, floatx8 c) {
  return __builtin_amdgcn_wmma_f32_16x16x32_bf16(false, a, false, b, (short)0, c,
                                                 false, false);
}

// ---------------- prep: fp32 weights -> bf16, gather rel-pos bias ----------------
__global__ void prep_kernel(const float* __restrict__ qkv_w,
                            const float* __restrict__ proj_w,
                            const float* __restrict__ bias_table,
                            const int* __restrict__ rel_index,
                            bf16_t* __restrict__ qkvw_b,
                            bf16_t* __restrict__ projw_b,
                            float* __restrict__ biasM) {
  int i = blockIdx.x * blockDim.x + threadIdx.x;
  if (i < QKVW_E)  qkvw_b[i] = (bf16_t)qkv_w[i];
  if (i < PROJW_E) projw_b[i] = (bf16_t)proj_w[i];
  if (i < BIAS_E) {
    int h = i >> 12;            // / 4096
    int nm = i & 4095;          // n*64+m
    biasM[i] = bias_table[rel_index[nm] * NHEAD + h];
  }
}

// ---------------- fused window attention, one block per window ----------------
// LDS tiles (one static __shared__ instance declared inside the kernel)
struct SharedMem {
  bf16_t xs[64][200];            // x tile (bf16), reused as attn output tile
  bf16_t qk_s[2][NHEAD][64][40]; // q (pre-scaled,+bias) and k, [tok][d]
  bf16_t vT_s[NHEAD][32][72];    // v transposed: [d][tok]
  bf16_t p_s[8][16][72];         // per-wave softmax(P) strip
};

// Phase B worker: 48 tiles for one of q/k/v (WHICH = 0/1/2), no data branch.
template <int WHICH>
__device__ __forceinline__ void qkv_phase(SharedMem& sm,
                                          const bf16_t* __restrict__ qkvw,
                                          const float* __restrict__ qkv_b,
                                          int wave, int half, int l16) {
  const float scale = 0.17677669529663687f;  // 32^-0.5
  for (int t = wave; t < 48; t += 8) {
    int jtl   = t >> 2;                // 0..11 within this third (SGPR math)
    int mtile = t & 3;
    int mbase = mtile * 16;
    int jbase = (WHICH * 12 + jtl) * 16;
    int h     = jtl >> 1;
    int dbase = (jtl & 1) * 16;
    int j     = jbase + l16;
    floatx8 acc = {};
#pragma unroll
    for (int kk = 0; kk < 6; ++kk) {
      int kb = kk * 32;
      bf16x16 a = frag_16(&sm.xs[mbase + l16][kb + half * 16]);
      bf16x16 bb = frag_16(qkvw + (size_t)j * CDIM + kb + half * 16);
      acc = wmma_bf16(a, bb, acc);
    }
    float bj = qkv_b[j];
    if (WHICH == 2) {
      // v: transpose-store, 8 consecutive tokens -> one b128 LDS store
      bf16x8 pk;
#pragma unroll
      for (int p = 0; p < 8; ++p) pk[p] = (bf16_t)(acc[p] + bj);
      *(bf16x8*)&sm.vT_s[h][dbase + l16][mbase + half * 8] = pk;
    } else {
      bf16_t* qp = &sm.qk_s[WHICH][h][mbase + half * 8][dbase + l16];
#pragma unroll
      for (int p = 0; p < 8; ++p) {
        float v = acc[p] + bj;
        if (WHICH == 0) v *= scale;
        qp[p * 40] = (bf16_t)v;       // immediate-offset ds_store
      }
    }
  }
}

__global__ __launch_bounds__(256, 1)
void win_attn_kernel(const float* __restrict__ x,
                     const float* __restrict__ mask,
                     const float* __restrict__ qkv_b,
                     const float* __restrict__ proj_b,
                     const bf16_t* __restrict__ qkvw,
                     const bf16_t* __restrict__ projw,
                     const float* __restrict__ biasM,
                     float* __restrict__ out) {
  __shared__ SharedMem sm;

  const int tid  = threadIdx.x;
  const int wave = __builtin_amdgcn_readfirstlane(tid >> 5);  // force SGPR
  const int lane = tid & 31;
  const int half = lane >> 4;
  const int l16  = lane & 15;
  const int b    = blockIdx.x;

  const float* maskb = mask + (size_t)(b & (NW - 1)) * 4096;
  // gfx1250 prefetch: pull the mask block and bias toward this WGP before Phase C
  __builtin_prefetch(maskb + tid * 16, 0, 0);   // 4096 f32 = 16 KB
  __builtin_prefetch(biasM + tid * 96, 0, 0);   // 24576 f32 = 96 KB (L2-resident)

  // ---- Phase A: stage x[b] (64x192 f32) -> bf16 LDS, packed b64 stores ----
  const float4* xb = (const float4*)(x + (size_t)b * NTOK * CDIM);
#pragma unroll
  for (int i = 0; i < 12; ++i) {
    int idx = tid + i * 256;       // 3072 float4 total
    float4 v = xb[idx];
    int e = idx * 4;
    int row = e / CDIM, col = e % CDIM;
    bf16x4 pk = { (bf16_t)v.x, (bf16_t)v.y, (bf16_t)v.z, (bf16_t)v.w };
    *(bf16x4*)&sm.xs[row][col] = pk;
  }
  __syncthreads();

  // ---- Phase B: qkv = x @ qkv_w^T + qkv_b (q, then k, then v: no branches) ----
  qkv_phase<0>(sm, qkvw, qkv_b, wave, half, l16);
  qkv_phase<1>(sm, qkvw, qkv_b, wave, half, l16);
  qkv_phase<2>(sm, qkvw, qkv_b, wave, half, l16);
  __syncthreads();

  // ---- Phase C: per (head, 16-row strip): S=qk^T, +bias+mask, softmax, P@V ----
  for (int s = wave; s < 24; s += 8) {
    int h = s >> 2, r = s & 3;
    int mbase = r * 16;
    floatx8 st[4];
#pragma unroll
    for (int ct = 0; ct < 4; ++ct) {
      bf16x16 a = frag_16(&sm.qk_s[0][h][mbase + l16][half * 16]);
      bf16x16 bb = frag_16(&sm.qk_s[1][h][ct * 16 + l16][half * 16]);
      floatx8 z = {};
      st[ct] = wmma_bf16(a, bb, z);
    }
    // base pointers; unrolled loops use immediate offsets only
    const float* bm = biasM + ((h * 64 + mbase + half * 8) * 64 + l16);
    const float* mk = maskb + ((mbase + half * 8) * 64 + l16);
    float rowm[8], rows[8];
#pragma unroll
    for (int p = 0; p < 8; ++p) rowm[p] = -3.4e38f;
#pragma unroll
    for (int ct = 0; ct < 4; ++ct) {
#pragma unroll
      for (int p = 0; p < 8; ++p) {
        float v = st[ct][p] + bm[p * 64 + ct * 16] + mk[p * 64 + ct * 16];
        st[ct][p] = v;
        rowm[p] = fmaxf(rowm[p], v);
      }
    }
#pragma unroll
    for (int off = 8; off >= 1; off >>= 1)
#pragma unroll
      for (int p = 0; p < 8; ++p)
        rowm[p] = fmaxf(rowm[p], __shfl_xor(rowm[p], off, 32));
#pragma unroll
    for (int p = 0; p < 8; ++p) rows[p] = 0.f;
#pragma unroll
    for (int ct = 0; ct < 4; ++ct)
#pragma unroll
      for (int p = 0; p < 8; ++p) {
        float e = __expf(st[ct][p] - rowm[p]);
        st[ct][p] = e;
        rows[p] += e;
      }
#pragma unroll
    for (int off = 8; off >= 1; off >>= 1)
#pragma unroll
      for (int p = 0; p < 8; ++p)
        rows[p] += __shfl_xor(rows[p], off, 32);
    // write unnormalized P (bf16) to this wave's strip (same-wave LDS RAW is in-order)
    bf16_t* pp = &sm.p_s[wave][half * 8][l16];
#pragma unroll
    for (int ct = 0; ct < 4; ++ct)
#pragma unroll
      for (int p = 0; p < 8; ++p)
        pp[p * 72 + ct * 16] = (bf16_t)st[ct][p];
    // P @ V ; divide by row-sum afterwards
    bf16_t* xo = &sm.xs[mbase + half * 8][h * HD + l16];
#pragma unroll
    for (int dt = 0; dt < 2; ++dt) {
      floatx8 o = {};
#pragma unroll
      for (int kt = 0; kt < 2; ++kt) {
        int kb = kt * 32;
        bf16x16 a = frag_16(&sm.p_s[wave][l16][kb + half * 16]);
        bf16x16 bb = frag_16(&sm.vT_s[h][dt * 16 + l16][kb + half * 16]);
        o = wmma_bf16(a, bb, o);
      }
#pragma unroll
      for (int p = 0; p < 8; ++p)
        xo[p * 200 + dt * 16] = (bf16_t)(o[p] / rows[p]);  // reuse xs as o-tile
    }
  }
  __syncthreads();

  // ---- Phase D: out = o @ proj_w^T + proj_b ; 4 Mtiles x 12 COtiles, K=192 ----
  float* ob = out + (size_t)b * NTOK * CDIM;
  for (int t = wave; t < 48; t += 8) {
    int cot = t >> 2, mtile = t & 3;        // SGPR math
    int mbase = mtile * 16, cobase = cot * 16;
    int co = cobase + l16;
    floatx8 acc = {};
#pragma unroll
    for (int kk = 0; kk < 6; ++kk) {
      int kb = kk * 32;
      bf16x16 a = frag_16(&sm.xs[mbase + l16][kb + half * 16]);
      bf16x16 bb = frag_16(projw + (size_t)co * CDIM + kb + half * 16);
      acc = wmma_bf16(a, bb, acc);
    }
    float bc = proj_b[co];
    float* obp = ob + (size_t)(mbase + half * 8) * CDIM + co;
#pragma unroll
    for (int p = 0; p < 8; ++p)
      obp[p * CDIM] = acc[p] + bc;          // immediate-offset global stores
  }
}

extern "C" void kernel_launch(void* const* d_in, const int* in_sizes, int n_in,
                              void* d_out, int out_size, void* d_ws, size_t ws_size,
                              hipStream_t stream) {
  const float* x      = (const float*)d_in[0];
  const float* mask   = (const float*)d_in[1];
  const float* qkv_w  = (const float*)d_in[2];
  const float* qkv_b  = (const float*)d_in[3];
  const float* proj_w = (const float*)d_in[4];
  const float* proj_b = (const float*)d_in[5];
  const float* bias_t = (const float*)d_in[6];
  const int*   rel_ix = (const int*)d_in[7];
  float* out = (float*)d_out;

  // workspace layout: [qkvw bf16 | projw bf16 | biasM f32]  (~384 KB)
  bf16_t* qkvw_b  = (bf16_t*)d_ws;
  bf16_t* projw_b = qkvw_b + QKVW_E;
  float*  biasM   = (float*)(projw_b + PROJW_E);

  prep_kernel<<<(QKVW_E + 255) / 256, 256, 0, stream>>>(
      qkv_w, proj_w, bias_t, rel_ix, qkvw_b, projw_b, biasM);
  win_attn_kernel<<<BWIN, 256, 0, stream>>>(
      x, mask, qkv_b, proj_b, qkvw_b, projw_b, biasM, out);
}
